// TopologicalModel_51488067945216
// MI455X (gfx1250) — matrix operand order repaired and checked by reading, not compiled, not run on hardware
//
#include <hip/hip_runtime.h>
#include <hip/hip_bf16.h>
#include <math.h>
#include <stdint.h>

// ---------------- types for WMMA ----------------
typedef __attribute__((ext_vector_type(16))) _Float16 v16h;
typedef __attribute__((ext_vector_type(8)))  _Float16 v8h;
typedef __attribute__((ext_vector_type(8)))  float    v8f;

union V16U { v16h v; v8h h[2]; };

// ---------------- TDM availability ----------------
#if defined(__has_builtin)
#  if __has_builtin(__builtin_amdgcn_tensor_load_to_lds) && __has_builtin(__builtin_amdgcn_s_wait_tensorcnt)
#    define TDM_AVAILABLE 1
#  else
#    define TDM_AVAILABLE 0
#  endif
#else
#  define TDM_AVAILABLE 0
#endif

#define MT 128
#define NT 64
#define KT 32
#define LDA_SH 40   // 32 + 8 halfs pad (80B row, keeps 16B alignment)
#define LDB_SH 40
#define MROWS 32768 // M is always B*S

#if TDM_AVAILABLE
typedef __attribute__((ext_vector_type(4))) unsigned int u32x4_t;
typedef __attribute__((ext_vector_type(8))) int          i32x8_t;
typedef __attribute__((ext_vector_type(4))) int          i32x4_t;

// Issue one TDM 2D tile load: 128 rows x 32 halfs from [MROWS,K] f16 tensor
// into LDS with 16B pad per 64B row (matches LDA_SH=40 halfs layout).
// Descriptor packing per CDNA5 ISA "Tensor DMA Descriptor (D#)" (8.3/8.4).
template<int K>
__device__ __forceinline__ void tdm_load_a_tile(const _Float16* gtile,
                                                unsigned int lds_byte_off)
{
    unsigned long long ga = (unsigned long long)(uintptr_t)gtile;
    u32x4_t g0;
    g0[0] = 1u;                                     // count=1, user descriptor
    g0[1] = lds_byte_off;                           // LDS dest byte address
    g0[2] = (unsigned int)ga;                       // global_addr[31:0]
    g0[3] = (unsigned int)(ga >> 32) | (2u << 30);  // global_addr[56:32] | type=2
    i32x8_t g1;
    g1[0] = (int)((1u << 16)      // data_size: 2 bytes
                | (1u << 20)      // pad_enable
                | (3u << 22)      // pad_interval: 16 DWORDs (64B tile row)
                | (3u << 25));    // pad_amount: 4 DWORDs (16B -> 80B LDS row)
    g1[1] = (int)(((unsigned)K & 0xFFFFu) << 16);                 // tensor_dim0 lo16
    g1[2] = (int)((((unsigned)K >> 16) & 0xFFFFu)
                | (((unsigned)MROWS & 0xFFFFu) << 16));           // dim0 hi | dim1 lo
    g1[3] = (int)((((unsigned)MROWS >> 16) & 0xFFFFu)
                | (32u << 16));                                   // dim1 hi | tile_dim0=32
    g1[4] = (int)128;                                             // tile_dim1=128, tile_dim2=0
    g1[5] = (int)K;                                               // tensor_dim0_stride lo32
    g1[6] = 0;                                                    // stride0 hi | stride1 lo
    g1[7] = 0;                                                    // stride1 hi
    i32x4_t gz = {0, 0, 0, 0};
#if defined(__clang_major__) && (__clang_major__ >= 23)
    i32x8_t gz8 = {0, 0, 0, 0, 0, 0, 0, 0};
    __builtin_amdgcn_tensor_load_to_lds(g0, g1, gz, gz, gz8, 0);
#else
    __builtin_amdgcn_tensor_load_to_lds(g0, g1, gz, gz, 0);
#endif
}
#endif // TDM_AVAILABLE

// =====================================================================
// Fused GEMM:  C = act( A[M,K] @ W[K,N] + bias[N] )
// Compile-time per-layer configuration.
// Block tile 128(M) x 64(N), 256 threads = 8 waves, wave tile 32x32.
// K stepped by 32 using v_wmma_f32_16x16x32_f16, f32 accumulate.
// AHALF layers stage A via the Tensor Data Mover (async DMA, TENSORcnt),
// double-buffered in LDS; fp32-A layers use VALU convert staging.
// =====================================================================
template<int K, int N, bool AHALF, bool ALPHA, bool CHALF, bool RELU>
__global__ __launch_bounds__(256) void gemm_wmma(
    const void* __restrict__ Ap,
    const float* __restrict__ alphas,
    const float* __restrict__ W,               // [K,N] row-major
    const float* __restrict__ bias,            // [N]
    void* __restrict__ Cout)
{
    constexpr int  KSTEPS  = (K + KT - 1) / KT;
    constexpr bool USE_TDM = AHALF && (TDM_AVAILABLE != 0);

    __shared__ _Float16 Ash[(USE_TDM ? 2 : 1) * MT * LDA_SH];
    __shared__ _Float16 Bsh[NT * LDB_SH];

    const int tid  = threadIdx.x;
    const int m0   = blockIdx.y * MT;
    const int n0   = blockIdx.x * NT;

    const int lane = tid & 31;
    const int lm   = lane & 15;
    const int hi   = lane >> 4;
    const int w    = tid >> 5;
    const int wm   = (w & 3) * 32;             // wave M offset in block tile
    const int wn   = (w >> 2) * 32;            // wave N offset in block tile

    // B staging geometry (constant across k-steps)
    const int be0  = tid * 8;
    const int bkk  = be0 >> 6;                 // 0..31
    const int bnnb = be0 & 63;                 // multiple of 8
    const int bgn0 = n0 + bnnb;

    // A staging geometry (VALU path)
    const int arow0 = tid >> 2,        akc0 = (tid & 3) * 8;
    const int arow1 = (tid + 256) >> 2, akc1 = ((tid + 256) & 3) * 8;

    v8f acc[2][2] = {};

#if TDM_AVAILABLE
    unsigned int ldsA0 = 0, ldsA1 = 0;
    const _Float16* Abase = nullptr;
    if constexpr (USE_TDM) {
        ldsA0 = (unsigned int)(uintptr_t)(void*)&Ash[0];
        ldsA1 = (unsigned int)(uintptr_t)(void*)&Ash[MT * LDA_SH];
        Abase = (const _Float16*)Ap + (size_t)m0 * K;
        if (w == 0) tdm_load_a_tile<K>(Abase, ldsA0);   // prologue: tile ks=0
    }
#endif

    for (int ks = 0; ks < KSTEPS; ++ks) {
        const int kb  = ks * KT;
        const int cur = USE_TDM ? (ks & 1) : 0;

        // ---- prefetch A into registers (VALU path only) ----
        _Float16 areg[2][8];
        if constexpr (!USE_TDM) {
            #pragma unroll
            for (int rep = 0; rep < 2; ++rep) {
                const int row = rep ? arow1 : arow0;
                const int kc  = rep ? akc1  : akc0;
                const int gk0 = kb + kc;
                const bool ok = (K % KT == 0) || (gk0 + 8 <= K);
                if (ok) {
                    const size_t base = (size_t)(m0 + row) * K + gk0;
                    if constexpr (AHALF) {
                        v8h v = *(const v8h*)((const _Float16*)Ap + base);
                        #pragma unroll
                        for (int j = 0; j < 8; ++j) areg[rep][j] = v[j];
                    } else {
                        const float* af = (const float*)Ap + base;
                        float4 f0 = *(const float4*)(af);
                        float4 f1 = *(const float4*)(af + 4);
                        float sc = 1.f;
                        if constexpr (ALPHA) sc = fabsf(alphas[gk0 >> 4]);
                        areg[rep][0] = (_Float16)(f0.x * sc);
                        areg[rep][1] = (_Float16)(f0.y * sc);
                        areg[rep][2] = (_Float16)(f0.z * sc);
                        areg[rep][3] = (_Float16)(f0.w * sc);
                        areg[rep][4] = (_Float16)(f1.x * sc);
                        areg[rep][5] = (_Float16)(f1.y * sc);
                        areg[rep][6] = (_Float16)(f1.z * sc);
                        areg[rep][7] = (_Float16)(f1.w * sc);
                    }
                } else {
                    #pragma unroll
                    for (int j = 0; j < 8; ++j) areg[rep][j] = (_Float16)0.f;
                }
            }
        }
        // ---- prefetch B (8 contiguous N elements of row kb+bkk) ----
        _Float16 breg[8];
        {
            const int gk = kb + bkk;
            const bool ok = ((K % KT == 0) || (gk < K)) &&
                            ((N % NT == 0) || (bgn0 + 8 <= N));
            if (ok) {
                const float* wf = W + (size_t)gk * N + bgn0;
                float4 f0 = *(const float4*)(wf);
                float4 f1 = *(const float4*)(wf + 4);
                breg[0] = (_Float16)f0.x; breg[1] = (_Float16)f0.y;
                breg[2] = (_Float16)f0.z; breg[3] = (_Float16)f0.w;
                breg[4] = (_Float16)f1.x; breg[5] = (_Float16)f1.y;
                breg[6] = (_Float16)f1.z; breg[7] = (_Float16)f1.w;
            } else {
                #pragma unroll
                for (int j = 0; j < 8; ++j) breg[j] = (_Float16)0.f;
            }
        }

        __syncthreads();   // previous iteration's LDS reads done

        // ---- store staged data to LDS ----
        if constexpr (!USE_TDM) {
            #pragma unroll
            for (int rep = 0; rep < 2; ++rep) {
                const int row = rep ? arow1 : arow0;
                const int kc  = rep ? akc1  : akc0;
                v8h v;
                #pragma unroll
                for (int j = 0; j < 8; ++j) v[j] = areg[rep][j];
                *(v8h*)&Ash[row * LDA_SH + kc] = v;        // 16B aligned
            }
        }
        #pragma unroll
        for (int j = 0; j < 8; ++j)
            Bsh[(bnnb + j) * LDB_SH + bkk] = breg[j];      // transposed scatter

#if TDM_AVAILABLE
        if constexpr (USE_TDM) {
            if (w == 0) {
                __builtin_amdgcn_s_wait_tensorcnt(0);      // tile ks landed
                if (ks + 1 < KSTEPS)                       // issue tile ks+1
                    tdm_load_a_tile<K>(Abase + (kb + KT), cur ? ldsA0 : ldsA1);
            }
        }
#endif
        __syncthreads();

        // ---- fragments (documented 16-bit A 16x32 / B 32x16 layouts) ----
        const _Float16* As = &Ash[cur * MT * LDA_SH];
        v16h afrag[2], bfrag[2];
        #pragma unroll
        for (int i = 0; i < 2; ++i) {
            const int r = wm + i * 16 + lm;
            V16U u;
            u.h[0] = *(const v8h*)&As[r * LDA_SH + hi * 8];
            u.h[1] = *(const v8h*)&As[r * LDA_SH + 16 + hi * 8];
            afrag[i] = u.v;
        }
        #pragma unroll
        for (int j = 0; j < 2; ++j) {
            const int n = wn + j * 16 + lm;
            V16U u;
            u.h[0] = *(const v8h*)&Bsh[n * LDB_SH + hi * 8];
            u.h[1] = *(const v8h*)&Bsh[n * LDB_SH + 16 + hi * 8];
            bfrag[j] = u.v;
        }

        #pragma unroll
        for (int i = 0; i < 2; ++i)
            #pragma unroll
            for (int j = 0; j < 2; ++j)
                acc[i][j] = __builtin_amdgcn_wmma_f32_16x16x32_f16(
                    false, afrag[i], false, bfrag[j],
                    (short)0, acc[i][j], false, false);
    }

    // ---- epilogue: bias + ReLU + store (N is a multiple of 16) ----
    #pragma unroll
    for (int j = 0; j < 2; ++j) {
        const int nb = n0 + wn + j * 16;
        if ((N % NT == 0) || (nb < N)) {       // whole 16-col tile valid
            const int gn = nb + lm;
            const float bb = bias[gn];
            #pragma unroll
            for (int i = 0; i < 2; ++i) {
                #pragma unroll
                for (int r = 0; r < 8; ++r) {
                    const int gm = m0 + wm + i * 16 + 8 * hi + r;
                    float v = acc[i][j][r] + bb;
                    if constexpr (RELU) v = fmaxf(v, 0.f);
                    if constexpr (CHALF)
                        ((_Float16*)Cout)[(size_t)gm * N + gn] = (_Float16)v;
                    else
                        ((float*)Cout)[(size_t)gm * N + gn] = v;
                }
            }
        }
    }
}

// =====================================================================
// Topo features: per batch item, normalize z[b] (unbiased column std),
// Prim-style loop EXACTLY as the reference writes it, 12 features out.
// One block per batch item, 512 threads (= S points).
// =====================================================================
#define S_PTS 512
#define LDIM  48

__device__ __forceinline__ float pair_dist(const _Float16* zn, const float* sq,
                                           int j, int t)
{
    float dot = 0.f;
    #pragma unroll
    for (int c = 0; c < LDIM; ++c)
        dot += (float)zn[j * LDIM + c] * (float)zn[t * LDIM + c];
    float d2 = sq[j] + sq[t] - 2.f * dot;
    return sqrtf(fmaxf(d2, 0.f));
}

__global__ __launch_bounds__(S_PTS) void topo_kernel(
    const float* __restrict__ zfull,   // [B, S, 48]
    float* __restrict__ topo_out)      // [B, 12]
{
    __shared__ _Float16 zn[S_PTS * LDIM];   // 48 KB
    __shared__ float sq[S_PTS];
    __shared__ float mind[S_PTS];
    __shared__ float rval[S_PTS];
    __shared__ int   ridx[S_PTS];
    __shared__ float cmean[LDIM];
    __shared__ float cstd[LDIM];

    const int t = threadIdx.x;
    const int b = blockIdx.x;
    const float* zb = zfull + (size_t)b * S_PTS * LDIM;

    // column mean / unbiased std
    if (t < LDIM) {
        float s = 0.f, s2 = 0.f;
        for (int i = 0; i < S_PTS; ++i) {
            float v = zb[i * LDIM + t];
            s += v; s2 += v * v;
        }
        float mean = s / (float)S_PTS;
        float var  = (s2 - (float)S_PTS * mean * mean) / (float)(S_PTS - 1);
        cmean[t] = mean;
        cstd[t]  = sqrtf(fmaxf(var, 0.f));
    }
    __syncthreads();

    // normalize into LDS (f16)
    for (int idx = t; idx < S_PTS * LDIM; idx += S_PTS) {
        int c = idx % LDIM;
        float v = (zb[idx] - cmean[c]) / (cstd[c] + 1e-8f);
        zn[idx] = (_Float16)v;
    }
    __syncthreads();

    // squared norms
    {
        float s = 0.f;
        #pragma unroll
        for (int c = 0; c < LDIM; ++c) {
            float v = (float)zn[t * LDIM + c];
            s += v * v;
        }
        sq[t] = s;
    }
    __syncthreads();

    // mind = D[0] with mind[0] = inf
    mind[t] = (t == 0) ? __builtin_inff() : pair_dist(zn, sq, 0, t);
    __syncthreads();

    float ssum = 0.f, ssq = 0.f, smax = 0.f;   // thread 0's copies are used

    for (int it = 0; it < S_PTS - 1; ++it) {
        rval[t] = mind[t];
        ridx[t] = t;
        __syncthreads();
        // argmin with first-index tie break (matches jnp.argmin)
        for (int off = S_PTS / 2; off > 0; off >>= 1) {
            if (t < off) {
                float o = rval[t + off]; int oi = ridx[t + off];
                if (o < rval[t] || (o == rval[t] && oi < ridx[t])) {
                    rval[t] = o; ridx[t] = oi;
                }
            }
            __syncthreads();
        }
        int   j = ridx[0];
        float wv = rval[0];
        if (t == 0) { ssum += wv; ssq += wv * wv; smax = fmaxf(smax, wv); }

        // mind = min(mind, D[j]); mind[j] = inf   (exact reference semantics)
        float dj = pair_dist(zn, sq, j, t);
        float nm = fminf(mind[t], dj);
        mind[t] = (t == j) ? __builtin_inff() : nm;
        __syncthreads();
    }

    if (t == 0) {
        const float n = (float)(S_PTS - 1);
        float mean = ssum / n;
        float var  = (ssq - n * mean * mean) / (n - 1.f);
        float sd   = sqrtf(fmaxf(var, 0.f));
        float* tp = topo_out + b * 12;
        tp[0] = mean; tp[1] = sd; tp[2] = smax;
        tp[3] = ssum; tp[4] = n;  tp[5] = smax;
        #pragma unroll
        for (int q = 6; q < 12; ++q) tp[q] = 0.f;
    }
}

// =====================================================================
// Classifier MLP: 12 -> 64 -> 32 -> 3 over 64 rows. One thread per row.
// =====================================================================
__global__ __launch_bounds__(64) void clf_kernel(
    const float* __restrict__ topo,
    const float* __restrict__ W1, const float* __restrict__ b1,
    const float* __restrict__ W2, const float* __restrict__ b2,
    const float* __restrict__ W3, const float* __restrict__ b3,
    float* __restrict__ logits)
{
    int r = threadIdx.x;                // 0..63
    float in[12];
    #pragma unroll
    for (int c = 0; c < 12; ++c) in[c] = topo[r * 12 + c];
    float h1[64];
    #pragma unroll
    for (int o = 0; o < 64; ++o) {
        float s = b1[o];
        #pragma unroll
        for (int c = 0; c < 12; ++c) s += in[c] * W1[c * 64 + o];
        h1[o] = fmaxf(s, 0.f);
    }
    float h2[32];
    #pragma unroll
    for (int o = 0; o < 32; ++o) {
        float s = b2[o];
        #pragma unroll
        for (int c = 0; c < 64; ++c) s += h1[c] * W2[c * 32 + o];
        h2[o] = fmaxf(s, 0.f);
    }
    #pragma unroll
    for (int o = 0; o < 3; ++o) {
        float s = b3[o];
        #pragma unroll
        for (int c = 0; c < 32; ++c) s += h2[c] * W3[c * 3 + o];
        logits[r * 3 + o] = s;
    }
}

// =====================================================================
// Host launcher
// =====================================================================
extern "C" void kernel_launch(void* const* d_in, const int* in_sizes, int n_in,
                              void* d_out, int out_size, void* d_ws, size_t ws_size,
                              hipStream_t stream)
{
    // setup_inputs() flattening order:
    //   x, node_alphas,
    //   encW1,encB1, encW2,encB2, encW3,encB3,
    //   decW1,decB1, decW2,decB2, decW3,decB3,
    //   clfW1,clfB1, clfW2,clfB2, clfW3,clfB3
    const float* x      = (const float*)d_in[0];
    const float* alphas = (const float*)d_in[1];
    const float* eW1 = (const float*)d_in[2];  const float* eB1 = (const float*)d_in[3];
    const float* eW2 = (const float*)d_in[4];  const float* eB2 = (const float*)d_in[5];
    const float* eW3 = (const float*)d_in[6];  const float* eB3 = (const float*)d_in[7];
    const float* dW1 = (const float*)d_in[8];  const float* dB1 = (const float*)d_in[9];
    const float* dW2 = (const float*)d_in[10]; const float* dB2 = (const float*)d_in[11];
    const float* dW3 = (const float*)d_in[12]; const float* dB3 = (const float*)d_in[13];
    const float* cW1 = (const float*)d_in[14]; const float* cB1 = (const float*)d_in[15];
    const float* cW2 = (const float*)d_in[16]; const float* cB2 = (const float*)d_in[17];
    const float* cW3 = (const float*)d_in[18]; const float* cB3 = (const float*)d_in[19];

    const int B = 64, S = 512, M = B * S;           // 32768 rows
    // d_out layout (flat, return order): logits, x_recon, z, topo
    float* out    = (float*)d_out;
    float* logits = out;                            // 64*3
    float* xrec   = out + 192;                      // 64*512*768
    float* z      = xrec + (size_t)M * 768;         // 64*512*48
    float* topo   = z    + (size_t)M * 48;          // 64*12

    // workspace: two f16 ping-pong buffers (peak 48 MB)
    char* wsb = (char*)d_ws;
    _Float16* bufA = (_Float16*)wsb;                           // up to 32768*512 f16 (32 MB)
    _Float16* bufB = (_Float16*)(wsb + (size_t)33554432);      // up to 32768*256 f16 (16 MB)

    const dim3 blk(256);
    const int gy = M / MT;                          // 256
    // encoder
    gemm_wmma<768, 512, false, true,  true,  true ><<<dim3(512/NT, gy), blk, 0, stream>>>(x,    alphas,  eW1, eB1, bufA);
    gemm_wmma<512, 256, true,  false, true,  true ><<<dim3(256/NT, gy), blk, 0, stream>>>(bufA, nullptr, eW2, eB2, bufB);
    gemm_wmma<256, 48,  true,  false, false, false><<<dim3(1,      gy), blk, 0, stream>>>(bufB, nullptr, eW3, eB3, z);
    // decoder
    gemm_wmma<48,  256, false, false, true,  true ><<<dim3(256/NT, gy), blk, 0, stream>>>(z,    nullptr, dW1, dB1, bufB);
    gemm_wmma<256, 512, true,  false, true,  true ><<<dim3(512/NT, gy), blk, 0, stream>>>(bufB, nullptr, dW2, dB2, bufA);
    gemm_wmma<512, 768, true,  false, false, false><<<dim3(768/NT, gy), blk, 0, stream>>>(bufA, nullptr, dW3, dB3, xrec);
    // topo features (reads z from d_out)
    topo_kernel<<<dim3(B), dim3(S_PTS), 0, stream>>>(z, topo);
    // classifier
    clf_kernel<<<dim3(1), dim3(64), 0, stream>>>(topo, cW1, cB1, cW2, cB2,
                                                 cW3, cB3, logits);
}
